// MLP_Discriminator_49357764166095
// MI455X (gfx1250) — matrix operand order repaired and checked by reading, not compiled, not run on hardware
//
#include <hip/hip_runtime.h>

typedef __attribute__((ext_vector_type(16))) _Float16 v16h;
typedef __attribute__((ext_vector_type(8)))  _Float16 v8h;
typedef __attribute__((ext_vector_type(8)))  float    v8f;

#define HID  256
#define NVAR 10
#define TT   64
#define BBAT 32
#define NE   90
#define R1   (BBAT*NVAR*TT)   /* 20480  node rows  */
#define R2   (BBAT*NE*TT)     /* 184320 edge rows  */

__device__ __forceinline__ float elu1(float x) {
  return x > 0.f ? x : (__expf(x) - 1.f);
}

// A-fragment (16-bit A 16x32): lane holds row (lane&15);
// halves [0..15] = K in [koff + hi*16, +16) (contiguous, matches B).
__device__ __forceinline__ v16h load_a_frag(const _Float16* rowp, int koff, int hi) {
  const _Float16* p = rowp + koff + hi * 16;
  v8h lo = *(const v8h*)(p);
  v8h up = *(const v8h*)(p + 8);
  v16h r;
#pragma unroll
  for (int i = 0; i < 8; ++i) { r[i] = lo[i]; r[i + 8] = up[i]; }
  return r;
}

__device__ __forceinline__ v8f wmma16(v16h a, v16h b, v8f c) {
  return __builtin_amdgcn_wmma_f32_16x16x32_f16(false, a, false, b, (short)0, c,
                                                false, false);
}

// One wave computes a 16x32 slab (2 WMMA n-tiles) of A[16xK] * Wt[256xK]^T.
// A in LDS row-major (stride K, halves). Wt is [out][in] (column-major B).
template <int K>
__device__ __forceinline__ void wave_fc(const _Float16* ldsA, const _Float16* wt,
                                        int lane, int wave, v8f& c0, v8f& c1) {
  const int hi = lane >> 4;
  const int m  = lane & 15;
  const _Float16* arow = ldsA + m * K;
  const _Float16* wq0  = wt + (size_t)(wave * 32 + m) * K;
  const _Float16* wq1  = wq0 + (size_t)16 * K;
  c0 = (v8f){}; c1 = (v8f){};
#pragma unroll
  for (int kk = 0; kk < K / 32; ++kk) {
    v16h a  = load_a_frag(arow, kk * 32, hi);
    v16h b0 = *(const v16h*)(wq0 + kk * 32 + hi * 16);
    v16h b1 = *(const v16h*)(wq1 + kk * 32 + hi * 16);
    c0 = wmma16(a, b0, c0);
    c1 = wmma16(a, b1, c1);
  }
}

// C/D layout: VGPR r, lane<16 -> row r, lane>=16 -> row r+8; col = (lane&15)+tilebase.
__device__ __forceinline__ void epi_lds(v8f c, int col, float bias, int hi,
                                        _Float16 (*lds)[HID]) {
#pragma unroll
  for (int r = 0; r < 8; ++r)
    lds[r + 8 * hi][col] = (_Float16)elu1(c[r] + bias);
}

__device__ __forceinline__ void epi_store(v8f c, int col, float bias, int hi, int base,
                                          _Float16* out, float* sum, float* sq) {
  float s = 0.f, q = 0.f;
#pragma unroll
  for (int r = 0; r < 8; ++r) {
    float v = elu1(c[r] + bias);
    out[(size_t)(base + r + 8 * hi) * HID + col] = (_Float16)v;
    s += v; q += v * v;
  }
  atomicAdd(&sum[col], s);
  atomicAdd(&sq[col], q);
}

// ---------------- weight transpose fp32[K,N] -> f16[N,K] ----------------
__global__ void k_cvt_t(const float* __restrict__ w, _Float16* __restrict__ wt,
                        int K, int N) {
  int idx = blockIdx.x * blockDim.x + threadIdx.x;
  if (idx >= K * N) return;
  int k = idx / N, n = idx - k * N;
  wt[(size_t)n * K + k] = (_Float16)w[idx];
}

__global__ void k_zero(float* p, int n) {
  int i = blockIdx.x * blockDim.x + threadIdx.x;
  if (i < n) p[i] = 0.f;
}

__global__ void k_finalize(const float* __restrict__ sum, const float* __restrict__ sq,
                           const float* __restrict__ gamma, const float* __restrict__ beta,
                           float* __restrict__ scale, float* __restrict__ shift, float cnt) {
  int c = threadIdx.x;
  float mean = sum[c] / cnt;
  float var  = sq[c] / cnt - mean * mean;
  float rstd = rsqrtf(var + 1e-5f);
  float sc   = gamma[c] * rstd;
  scale[c] = sc;
  shift[c] = beta[c] - mean * sc;
}

// ---------------- mlp1: x[B,T,N,4] -> h2[R1,256] (elu'd, pre-BN) ----------------
__global__ void __launch_bounds__(256)
k_mlp1(const float* __restrict__ x, const float* __restrict__ w1a,
       const float* __restrict__ b1a, const _Float16* __restrict__ w1bt,
       const float* __restrict__ b1b, _Float16* __restrict__ h2,
       float* __restrict__ sum1, float* __restrict__ sq1) {
  __shared__ __align__(32) _Float16 ldsH[16][HID];
  const int tid = threadIdx.x;
  const int base = blockIdx.x * 16;

  { // fc1, K=4: each thread owns one output column for all 16 rows
    int col = tid;
    float wv0 = w1a[0 * HID + col], wv1 = w1a[1 * HID + col];
    float wv2 = w1a[2 * HID + col], wv3 = w1a[3 * HID + col];
    float bia = b1a[col];
#pragma unroll 4
    for (int r = 0; r < 16; ++r) {
      int g = base + r;
      int b = g / (NVAR * TT);
      int rem = g - b * (NVAR * TT);
      int n = rem / TT;
      int t = rem - n * TT;
      const float* xp = x + (((size_t)(b * TT + t)) * NVAR + n) * 4;
      float v = xp[0] * wv0 + xp[1] * wv1 + xp[2] * wv2 + xp[3] * wv3 + bia;
      ldsH[r][col] = (_Float16)elu1(v);
    }
  }
  __syncthreads();

  const int lane = tid & 31, wave = tid >> 5, hi = lane >> 4;
  v8f c0, c1;
  wave_fc<HID>(&ldsH[0][0], w1bt, lane, wave, c0, c1);
  int col = wave * 32 + (lane & 15);
  epi_store(c0, col,      b1b[col],      hi, base, h2, sum1, sq1);
  epi_store(c1, col + 16, b1b[col + 16], hi, base, h2, sum1, sq1);
}

// ------------- mlp2: gather(send||recv) + BN1 -> 512->256->256 -------------
__global__ void __launch_bounds__(256)
k_mlp2(const _Float16* __restrict__ h2, const float* __restrict__ sc1,
       const float* __restrict__ sh1, const _Float16* __restrict__ w2at,
       const float* __restrict__ b2a, const _Float16* __restrict__ w2bt,
       const float* __restrict__ b2b, _Float16* __restrict__ e2,
       float* __restrict__ sum2, float* __restrict__ sq2) {
  __shared__ __align__(32) _Float16 ldsA[16][2 * HID];
  __shared__ __align__(32) _Float16 ldsH[16][HID];
  const int tid = threadIdx.x;
  const int base = blockIdx.x * 16;

  { // stage A: per-thread 32 contiguous halves of the gathered, BN1'd 16x512 tile
    int row = tid >> 4;
    int chunk = tid & 15;                 // 16 chunks of 32 cols
    int g = base + row;
    int b = g / (NE * TT);
    int rem = g - b * (NE * TT);
    int e = rem / TT;
    int t = rem - e * TT;
    int s = e / 9;
    int w9 = e - s * 9;
    int rv = w9 + (w9 >= s ? 1 : 0);      // receiver node
    int node = (chunk < 8) ? s : rv;      // first 256 cols = sender, next 256 = receiver
    int srcRow = (b * NVAR + node) * TT + t;
    int c0 = (chunk & 7) * 32;            // channel within h
    const _Float16* src = h2 + (size_t)srcRow * HID + c0;
    _Float16* dst = &ldsA[row][chunk * 32];
#pragma unroll
    for (int j = 0; j < 32; ++j) {
      float v = (float)src[j];
      int ch = c0 + j;
      dst[j] = (_Float16)(v * sc1[ch] + sh1[ch]);
    }
  }
  __syncthreads();

  const int lane = tid & 31, wave = tid >> 5, hi = lane >> 4;
  v8f c0, c1;
  wave_fc<2 * HID>(&ldsA[0][0], w2at, lane, wave, c0, c1);
  int col = wave * 32 + (lane & 15);
  epi_lds(c0, col,      b2a[col],      hi, ldsH);
  epi_lds(c1, col + 16, b2a[col + 16], hi, ldsH);
  __syncthreads();
  wave_fc<HID>(&ldsH[0][0], w2bt, lane, wave, c0, c1);
  epi_store(c0, col,      b2b[col],      hi, base, e2, sum2, sq2);
  epi_store(c1, col + 16, b2b[col + 16], hi, base, e2, sum2, sq2);
}

// ------------- edge2node: BN2 (affine, commutes with sum) + /9 -------------
__global__ void k_edge2node(const _Float16* __restrict__ e2,
                            const float* __restrict__ sc2, const float* __restrict__ sh2,
                            _Float16* __restrict__ nin) {
  int idx = blockIdx.x * blockDim.x + threadIdx.x;  // R1*32 threads, 8 chans each
  int row = idx >> 5;
  if (row >= R1) return;
  int c0 = (idx & 31) * 8;
  int b = row / (NVAR * TT);
  int rem = row - b * (NVAR * TT);
  int v = rem / TT;
  int t = rem - v * TT;
  float acc[8] = {0.f, 0.f, 0.f, 0.f, 0.f, 0.f, 0.f, 0.f};
  for (int s = 0; s < NVAR; ++s) {
    if (s == v) continue;
    int w9 = (v < s) ? v : v - 1;         // edge with sender s, receiver v
    int e = s * 9 + w9;
    const _Float16* src = e2 + ((size_t)((b * NE + e) * TT + t)) * HID + c0;
#pragma unroll
    for (int j = 0; j < 8; ++j) acc[j] += (float)src[j];
  }
  const float inv9 = 1.f / 9.f;
#pragma unroll
  for (int j = 0; j < 8; ++j) {
    int ch = c0 + j;
    nin[(size_t)row * HID + ch] =
        (_Float16)((acc[j] * sc2[ch] + 9.f * sh2[ch]) * inv9);
  }
}

// ---------------- mlp3: nin[R1,256] -> 256 -> 256 (elu, pre-BN) ----------------
__global__ void __launch_bounds__(256)
k_mlp_mid(const _Float16* __restrict__ in, const _Float16* __restrict__ wat,
          const float* __restrict__ ba, const _Float16* __restrict__ wbt,
          const float* __restrict__ bb, _Float16* __restrict__ out,
          float* __restrict__ sum, float* __restrict__ sq) {
  __shared__ __align__(32) _Float16 ldsA[16][HID];
  __shared__ __align__(32) _Float16 ldsH[16][HID];
  const int tid = threadIdx.x;
  const int base = blockIdx.x * 16;
  {
    int row = tid >> 4, c0 = (tid & 15) * 16;
    *(v16h*)&ldsA[row][c0] = *(const v16h*)(in + (size_t)(base + row) * HID + c0);
  }
  __syncthreads();
  const int lane = tid & 31, wave = tid >> 5, hi = lane >> 4;
  v8f c0, c1;
  wave_fc<HID>(&ldsA[0][0], wat, lane, wave, c0, c1);
  int col = wave * 32 + (lane & 15);
  epi_lds(c0, col,      ba[col],      hi, ldsH);
  epi_lds(c1, col + 16, ba[col + 16], hi, ldsH);
  __syncthreads();
  wave_fc<HID>(&ldsH[0][0], wbt, lane, wave, c0, c1);
  epi_store(c0, col,      bb[col],      hi, base, out, sum, sq);
  epi_store(c1, col + 16, bb[col + 16], hi, base, out, sum, sq);
}

// ---------- head: relu(BN3) @ wo1 + bo1, relu, @ wo2 + bo2, transpose ----------
__global__ void __launch_bounds__(256)
k_head(const _Float16* __restrict__ n2, const float* __restrict__ sc3,
       const float* __restrict__ sh3, const _Float16* __restrict__ wo1t,
       const float* __restrict__ bo1, const float* __restrict__ wo2,
       const float* __restrict__ bo2, float* __restrict__ out) {
  __shared__ __align__(32) _Float16 ldsA[16][HID];
  __shared__ float ldsO[16][HID];
  const int tid = threadIdx.x;
  const int base = blockIdx.x * 16;
  {
    int row = tid >> 4, c0 = (tid & 15) * 16;
    const _Float16* src = n2 + (size_t)(base + row) * HID + c0;
#pragma unroll
    for (int j = 0; j < 16; ++j) {
      float v = (float)src[j] * sc3[c0 + j] + sh3[c0 + j];
      ldsA[row][c0 + j] = (_Float16)fmaxf(v, 0.f);
    }
  }
  __syncthreads();
  const int lane = tid & 31, wave = tid >> 5, hi = lane >> 4;
  v8f c0, c1;
  wave_fc<HID>(&ldsA[0][0], wo1t, lane, wave, c0, c1);
#pragma unroll
  for (int nt = 0; nt < 2; ++nt) {
    int col = wave * 32 + nt * 16 + (lane & 15);
    float bias = bo1[col];
#pragma unroll
    for (int r = 0; r < 8; ++r) {
      float v = (nt ? c1[r] : c0[r]) + bias;
      ldsO[r + 8 * hi][col] = fmaxf(v, 0.f);
    }
  }
  __syncthreads();
  if (tid < 16) {                          // 256-dot with wo2 per row + transpose
    float acc = bo2[0];
    for (int k = 0; k < HID; ++k) acc += ldsO[tid][k] * wo2[k];
    int g = base + tid;
    int b = g / (NVAR * TT);
    int rem = g - b * (NVAR * TT);
    int v = rem / TT;
    int t = rem - v * TT;
    out[((size_t)(b * TT + t)) * NVAR + v] = acc;  // [B,T,N]
  }
}

extern "C" void kernel_launch(void* const* d_in, const int* in_sizes, int n_in,
                              void* d_out, int out_size, void* d_ws, size_t ws_size,
                              hipStream_t stream) {
  (void)in_sizes; (void)n_in; (void)out_size; (void)ws_size;
  const float* x   = (const float*)d_in[0];
  const float* w1a = (const float*)d_in[1];
  const float* b1a = (const float*)d_in[2];
  const float* w1b = (const float*)d_in[3];
  const float* b1b = (const float*)d_in[4];
  const float* g1  = (const float*)d_in[5];
  const float* be1 = (const float*)d_in[6];
  const float* w2a = (const float*)d_in[7];
  const float* b2a = (const float*)d_in[8];
  const float* w2b = (const float*)d_in[9];
  const float* b2b = (const float*)d_in[10];
  const float* g2  = (const float*)d_in[11];
  const float* be2 = (const float*)d_in[12];
  const float* w3a = (const float*)d_in[13];
  const float* b3a = (const float*)d_in[14];
  const float* w3b = (const float*)d_in[15];
  const float* b3b = (const float*)d_in[16];
  const float* g3  = (const float*)d_in[17];
  const float* be3 = (const float*)d_in[18];
  const float* wo1 = (const float*)d_in[19];
  const float* bo1 = (const float*)d_in[20];
  const float* wo2 = (const float*)d_in[21];
  const float* bo2 = (const float*)d_in[22];

  char* ws = (char*)d_ws;
  size_t off = 0;
  auto alloc = [&](size_t bytes) -> void* {
    void* p = ws + off;
    off = (off + bytes + 255) & ~(size_t)255;
    return p;
  };

  _Float16* w1bt = (_Float16*)alloc((size_t)HID * HID * 2);
  _Float16* w2at = (_Float16*)alloc((size_t)2 * HID * HID * 2);
  _Float16* w2bt = (_Float16*)alloc((size_t)HID * HID * 2);
  _Float16* w3at = (_Float16*)alloc((size_t)HID * HID * 2);
  _Float16* w3bt = (_Float16*)alloc((size_t)HID * HID * 2);
  _Float16* wo1t = (_Float16*)alloc((size_t)HID * HID * 2);
  float* stats   = (float*)alloc((size_t)12 * HID * 4);
  float *sum1 = stats + 0 * HID, *sq1 = stats + 1 * HID;
  float *sum2 = stats + 2 * HID, *sq2 = stats + 3 * HID;
  float *sum3 = stats + 4 * HID, *sq3 = stats + 5 * HID;
  float *sc1 = stats + 6 * HID,  *sh1 = stats + 7 * HID;
  float *sc2 = stats + 8 * HID,  *sh2 = stats + 9 * HID;
  float *sc3 = stats + 10 * HID, *sh3 = stats + 11 * HID;
  _Float16* h2  = (_Float16*)alloc((size_t)R1 * HID * 2);
  _Float16* e2  = (_Float16*)alloc((size_t)R2 * HID * 2);
  _Float16* nin = (_Float16*)alloc((size_t)R1 * HID * 2);
  _Float16* n2b = (_Float16*)alloc((size_t)R1 * HID * 2);

  k_zero<<<(6 * HID + 255) / 256, 256, 0, stream>>>(stats, 6 * HID);
  k_cvt_t<<<(HID * HID + 255) / 256, 256, 0, stream>>>(w1b, w1bt, HID, HID);
  k_cvt_t<<<(2 * HID * HID + 255) / 256, 256, 0, stream>>>(w2a, w2at, 2 * HID, HID);
  k_cvt_t<<<(HID * HID + 255) / 256, 256, 0, stream>>>(w2b, w2bt, HID, HID);
  k_cvt_t<<<(HID * HID + 255) / 256, 256, 0, stream>>>(w3a, w3at, HID, HID);
  k_cvt_t<<<(HID * HID + 255) / 256, 256, 0, stream>>>(w3b, w3bt, HID, HID);
  k_cvt_t<<<(HID * HID + 255) / 256, 256, 0, stream>>>(wo1, wo1t, HID, HID);

  k_mlp1<<<R1 / 16, 256, 0, stream>>>(x, w1a, b1a, w1bt, b1b, h2, sum1, sq1);
  k_finalize<<<1, 256, 0, stream>>>(sum1, sq1, g1, be1, sc1, sh1, (float)R1);
  k_mlp2<<<R2 / 16, 256, 0, stream>>>(h2, sc1, sh1, w2at, b2a, w2bt, b2b, e2, sum2, sq2);
  k_finalize<<<1, 256, 0, stream>>>(sum2, sq2, g2, be2, sc2, sh2, (float)R2);
  k_edge2node<<<(R1 * 32) / 256, 256, 0, stream>>>(e2, sc2, sh2, nin);
  k_mlp_mid<<<R1 / 16, 256, 0, stream>>>(nin, w3at, b3a, w3bt, b3b, n2b, sum3, sq3);
  k_finalize<<<1, 256, 0, stream>>>(sum3, sq3, g3, be3, sc3, sh3, (float)R1);
  k_head<<<R1 / 16, 256, 0, stream>>>(n2b, sc3, sh3, wo1t, bo1, wo2, bo2, (float*)d_out);
}
